// LSTMModel_45887430590853
// MI455X (gfx1250) — matrix operand order repaired and checked by reading, not compiled, not run on hardware
//
#include <hip/hip_runtime.h>
#include <hip/hip_bf16.h>

// ---------------- problem constants ----------------
#define BB   256
#define TT   1024
#define II   32
#define HH   128
#define GG   512          // 4*H
#define GBS  516          // gbuf row stride (floats): 516 % 8 == 4 -> no LDS bank aliasing between M and M+8 rows
#define MB   16           // batch rows per workgroup
#define NWG  (BB / MB)    // 16 workgroups
#define NTILES (GG / 16)  // 32 N-tiles of 16 gate columns
#define KC_H   (HH / 32)  // 4 K-chunks of 32 for K=128

typedef __attribute__((ext_vector_type(16))) __bf16 v16bf;
typedef __attribute__((ext_vector_type(8)))  float  v8f;

union FragB { uint4 q[2]; v16bf v; unsigned short us[16]; };
union Pack8 { uint4 q; unsigned short us[8]; };

// cold-path (pack kernels): explicit RNE
static __device__ __forceinline__ unsigned short f32_to_bf16_rne(float f) {
    unsigned int u = __float_as_uint(f);
    u = (u + 0x7FFFu + ((u >> 16) & 1u)) >> 16;
    return (unsigned short)u;
}
// hot-path: hardware convert
static __device__ __forceinline__ unsigned short bf16_bits(float x) {
    __bf16 b = (__bf16)x;
    return __builtin_bit_cast(unsigned short, b);
}
static __device__ __forceinline__ float sigm(float x) {
    return 1.0f / (1.0f + __expf(-x));
}
static __device__ __forceinline__ float tanh_fast(float x) {
    float e = __expf(2.0f * x);
    return 1.0f - 2.0f / (e + 1.0f);
}

// ---- A-fragment (16x32 bf16) from an LDS row-major [16][128] bf16 buffer ----
// lane<16 : row m=lane,   slots0-7 = K kc*32+0..7,  slots8-15 = K kc*32+16..23
// lane>=16: row m=lane-16,slots0-7 = K kc*32+8..15, slots8-15 = K kc*32+24..31
static __device__ __forceinline__ v16bf load_a_lds(const unsigned short* hbuf, int lane, int kc) {
    int m  = lane & 15;
    int kb = (lane >> 4) * 8;
    const unsigned short* p = hbuf + m * HH + kc * 32 + kb;
    FragB f;
    f.q[0] = *(const uint4*)(p);
    f.q[1] = *(const uint4*)(p + 16);
    return f.v;
}

// ---- B-fragment (32x16 bf16), packed: 32 lanes x 16 contiguous bf16 ----
static __device__ __forceinline__ v16bf load_b_packed(const unsigned short* base, int blk, int lane) {
    const unsigned short* p = base + ((size_t)blk * 32 + lane) * 16;
    FragB f;
    f.q[0] = *(const uint4*)(p);
    f.q[1] = *(const uint4*)(p + 8);
    return f.v;
}

// ---- A-fragment of x_t (fp32 global, K=I=32), hw bf16 converts ----
static __device__ __forceinline__ v16bf load_a_x(const float* __restrict__ x,
                                                 int b0, int t, int lane) {
    int m  = lane & 15;
    int kb = (lane >> 4) * 8;
    const float* p = x + ((size_t)(b0 + m) * TT + t) * II + kb;
    float4 f0 = *(const float4*)(p);
    float4 f1 = *(const float4*)(p + 4);
    float4 g0 = *(const float4*)(p + 16);
    float4 g1 = *(const float4*)(p + 20);
    FragB f;
    f.us[0]  = bf16_bits(f0.x); f.us[1]  = bf16_bits(f0.y);
    f.us[2]  = bf16_bits(f0.z); f.us[3]  = bf16_bits(f0.w);
    f.us[4]  = bf16_bits(f1.x); f.us[5]  = bf16_bits(f1.y);
    f.us[6]  = bf16_bits(f1.z); f.us[7]  = bf16_bits(f1.w);
    f.us[8]  = bf16_bits(g0.x); f.us[9]  = bf16_bits(g0.y);
    f.us[10] = bf16_bits(g0.z); f.us[11] = bf16_bits(g0.w);
    f.us[12] = bf16_bits(g1.x); f.us[13] = bf16_bits(g1.y);
    f.us[14] = bf16_bits(g1.z); f.us[15] = bf16_bits(g1.w);
    return f.v;
}

#define WMMA_BF16(A, B, C) \
    __builtin_amdgcn_wmma_f32_16x16x32_bf16(false, (A), false, (B), (short)0, (C), false, false)

// ================= weight packing pre-pass =================
// dst layout: [nt(32)][kc(nkc)][lane(32)][s(16)]   (B-fragment order)
// value = w[g][k],  g = nt*16 + (lane&15),  k = kc*32 + (lane>>4)*8 + (s>>3)*16 + (s&7)
__global__ void pack_b_kernel(const float* __restrict__ w, unsigned short* __restrict__ dst,
                              int K, int nkc) {
    int idx = blockIdx.x * blockDim.x + threadIdx.x;
    int total = NTILES * nkc * 512;
    if (idx >= total) return;
    int s    = idx & 15;
    int lane = (idx >> 4) & 31;
    int t2   = idx >> 9;              // = nt*nkc + kc
    int kc   = t2 % nkc;
    int nt   = t2 / nkc;
    int g = nt * 16 + (lane & 15);
    int k = kc * 32 + ((lane >> 4) << 3) + ((s & 8) << 1) + (s & 7);
    dst[idx] = f32_to_bf16_rne(w[(size_t)g * K + k]);
}

__global__ void bias_sum_kernel(const float* __restrict__ a, const float* __restrict__ b,
                                float* __restrict__ dst) {
    int i = blockIdx.x * blockDim.x + threadIdx.x;
    if (i < GG) dst[i] = a[i] + b[i];
}

// ================= fused 2-layer persistent LSTM =================
// grid = 16 workgroups (one per 16-row batch tile), block = 256 threads = 8 waves.
// LDS: w_hh0(128K) + w_hh1(128K) + gbuf(33K) + h1(4K) + h2(4K) + bias(4K) ~= 300KB
__global__ void
__launch_bounds__(256, 1)
lstm_fused_kernel(const float* __restrict__ x,
                  const unsigned short* __restrict__ pw_ih0,
                  const unsigned short* __restrict__ pw_hh0,
                  const unsigned short* __restrict__ pw_ih1,
                  const unsigned short* __restrict__ pw_hh1,
                  const float* __restrict__ bias0,
                  const float* __restrict__ bias1,
                  float* __restrict__ h2last) {
    extern __shared__ char smem[];
    unsigned short* whh0_l = (unsigned short*)(smem);                 // 131072 B
    unsigned short* whh1_l = (unsigned short*)(smem + 131072);        // 131072 B
    float*          gbuf   = (float*)(smem + 262144);                 //  33024 B (16 x 516 f32)
    unsigned short* h1_l   = (unsigned short*)(smem + 295168);        //   4096 B
    unsigned short* h2_l   = (unsigned short*)(smem + 299264);        //   4096 B
    float*          b0_l   = (float*)(smem + 303360);                 //   2048 B
    float*          b1_l   = (float*)(smem + 305408);                 //   2048 B
                                                                      // total 307456 B

    const int tid  = threadIdx.x;
    const int lane = tid & 31;
    const int wid  = tid >> 5;
    const int b0   = blockIdx.x * MB;

    // ---- stage resident weights into LDS ----
    {
        const uint4* s0 = (const uint4*)pw_hh0;
        const uint4* s1 = (const uint4*)pw_hh1;
        uint4* d0 = (uint4*)whh0_l;
        uint4* d1 = (uint4*)whh1_l;
        #pragma unroll 4
        for (int i = tid; i < 8192; i += 256) { d0[i] = s0[i]; d1[i] = s1[i]; }
        for (int i = tid; i < GG; i += 256) { b0_l[i] = bias0[i]; b1_l[i] = bias1[i]; }
        for (int i = tid; i < MB * HH / 8; i += 256) {
            ((uint4*)h1_l)[i] = make_uint4(0, 0, 0, 0);
            ((uint4*)h2_l)[i] = make_uint4(0, 0, 0, 0);
        }
    }
    // ---- cache w_ih0 B-fragments in registers (K=32 -> 1 chunk, 4 N-tiles/wave) ----
    v16bf wih0f[4];
    #pragma unroll
    for (int i = 0; i < 4; ++i)
        wih0f[i] = load_b_packed(pw_ih0, (wid * 4 + i), lane);

    __syncthreads();

    // per-thread cell state: thread owns row bb = tid>>4, cols j0..j0+7
    const int bb = tid >> 4;
    const int j0 = (tid & 15) * 8;
    float c1[8], c2[8];
    #pragma unroll
    for (int k = 0; k < 8; ++k) { c1[k] = 0.0f; c2[k] = 0.0f; }

    const int nlo  = lane & 15;
    const int mrow = (lane >> 4) * 8;
    const float* gbe = gbuf + bb * GBS + j0;   // elementwise base

    // prefetch base for x (per-lane row pointer)
    const float* xrow = x + ((size_t)(b0 + (lane & 15)) * TT) * II + (lane >> 4) * 8;

    for (int t = 0; t < TT; ++t) {
        // prefetch next timestep's x straight into cache hierarchy
        if (t + 1 < TT) __builtin_prefetch(xrow + (size_t)(t + 1) * II, 0, 0);

        // ================= layer 0: gates = x_t@Wih0^T + h1@Whh0^T + b =================
        v16bf a_x = load_a_x(x, b0, t, lane);
        v16bf a1[KC_H];
        #pragma unroll
        for (int kc = 0; kc < KC_H; ++kc) a1[kc] = load_a_lds(h1_l, lane, kc);

        #pragma unroll
        for (int i = 0; i < 4; ++i) {
            int nt = wid * 4 + i;
            float bia = b0_l[nt * 16 + nlo];
            v8f acc = {bia, bia, bia, bia, bia, bia, bia, bia};
            acc = WMMA_BF16(a_x, wih0f[i], acc);
            #pragma unroll
            for (int kc = 0; kc < KC_H; ++kc) {
                v16bf bh = load_b_packed(whh0_l, nt * KC_H + kc, lane);
                acc = WMMA_BF16(a1[kc], bh, acc);
            }
            #pragma unroll
            for (int r = 0; r < 8; ++r)
                gbuf[(mrow + r) * GBS + nt * 16 + nlo] = acc[r];
        }
        __syncthreads();

        // ---- layer 0 elementwise (vector LDS reads: 8 x ds_load_b128) ----
        {
            float4 iA = *(const float4*)(gbe);              float4 iB = *(const float4*)(gbe + 4);
            float4 fA = *(const float4*)(gbe + HH);         float4 fB = *(const float4*)(gbe + HH + 4);
            float4 gA = *(const float4*)(gbe + 2 * HH);     float4 gB = *(const float4*)(gbe + 2 * HH + 4);
            float4 oA = *(const float4*)(gbe + 3 * HH);     float4 oB = *(const float4*)(gbe + 3 * HH + 4);
            float ig[8] = {iA.x, iA.y, iA.z, iA.w, iB.x, iB.y, iB.z, iB.w};
            float fg[8] = {fA.x, fA.y, fA.z, fA.w, fB.x, fB.y, fB.z, fB.w};
            float gg[8] = {gA.x, gA.y, gA.z, gA.w, gB.x, gB.y, gB.z, gB.w};
            float og[8] = {oA.x, oA.y, oA.z, oA.w, oB.x, oB.y, oB.z, oB.w};
            Pack8 hp;
            #pragma unroll
            for (int k = 0; k < 8; ++k) {
                float c = sigm(fg[k]) * c1[k] + sigm(ig[k]) * tanh_fast(gg[k]);
                c1[k] = c;
                float h = sigm(og[k]) * tanh_fast(c);
                hp.us[k] = bf16_bits(h);
            }
            *(uint4*)(h1_l + bb * HH + j0) = hp.q;
        }
        __syncthreads();

        // ================= layer 1: gates = h1_t@Wih1^T + h2@Whh1^T + b =================
        v16bf a1n[KC_H], a2[KC_H];
        #pragma unroll
        for (int kc = 0; kc < KC_H; ++kc) {
            a1n[kc] = load_a_lds(h1_l, lane, kc);
            a2[kc]  = load_a_lds(h2_l, lane, kc);
        }
        #pragma unroll
        for (int i = 0; i < 4; ++i) {
            int nt = wid * 4 + i;
            float bia = b1_l[nt * 16 + nlo];
            v8f acc = {bia, bia, bia, bia, bia, bia, bia, bia};
            #pragma unroll
            for (int kc = 0; kc < KC_H; ++kc) {
                v16bf bi = load_b_packed(pw_ih1, nt * KC_H + kc, lane);  // streams from L2
                acc = WMMA_BF16(a1n[kc], bi, acc);
                v16bf bh = load_b_packed(whh1_l, nt * KC_H + kc, lane);
                acc = WMMA_BF16(a2[kc], bh, acc);
            }
            #pragma unroll
            for (int r = 0; r < 8; ++r)
                gbuf[(mrow + r) * GBS + nt * 16 + nlo] = acc[r];
        }
        __syncthreads();

        // ---- layer 1 elementwise ----
        {
            float4 iA = *(const float4*)(gbe);              float4 iB = *(const float4*)(gbe + 4);
            float4 fA = *(const float4*)(gbe + HH);         float4 fB = *(const float4*)(gbe + HH + 4);
            float4 gA = *(const float4*)(gbe + 2 * HH);     float4 gB = *(const float4*)(gbe + 2 * HH + 4);
            float4 oA = *(const float4*)(gbe + 3 * HH);     float4 oB = *(const float4*)(gbe + 3 * HH + 4);
            float ig[8] = {iA.x, iA.y, iA.z, iA.w, iB.x, iB.y, iB.z, iB.w};
            float fg[8] = {fA.x, fA.y, fA.z, fA.w, fB.x, fB.y, fB.z, fB.w};
            float gg[8] = {gA.x, gA.y, gA.z, gA.w, gB.x, gB.y, gB.z, gB.w};
            float og[8] = {oA.x, oA.y, oA.z, oA.w, oB.x, oB.y, oB.z, oB.w};
            Pack8 hp;
            #pragma unroll
            for (int k = 0; k < 8; ++k) {
                float c = sigm(fg[k]) * c2[k] + sigm(ig[k]) * tanh_fast(gg[k]);
                c2[k] = c;
                float h = sigm(og[k]) * tanh_fast(c);
                hp.us[k] = bf16_bits(h);
                if (t == TT - 1) h2last[(size_t)(b0 + bb) * HH + j0 + k] = h;
            }
            *(uint4*)(h2_l + bb * HH + j0) = hp.q;
        }
        __syncthreads();
    }
}

// ================= final FC heads =================
__global__ void fc_kernel(const float* __restrict__ h2last,
                          const float* __restrict__ fc1w, const float* __restrict__ fc1b,
                          const float* __restrict__ fc2w, const float* __restrict__ fc2b,
                          float* __restrict__ out) {
    __shared__ float hrow[HH];
    int b = blockIdx.x;
    int tid = threadIdx.x;   // 128 threads
    hrow[tid] = h2last[(size_t)b * HH + tid];
    __syncthreads();
    if (tid < 27) {
        float s = fc1b[tid];
        #pragma unroll 8
        for (int k = 0; k < HH; ++k) s = fmaf(hrow[k], fc1w[tid * HH + k], s);
        out[(size_t)b * 27 + tid] = s;
    } else if (tid == 27) {
        float s = fc2b[0];
        #pragma unroll 8
        for (int k = 0; k < HH; ++k) s = fmaf(hrow[k], fc2w[k], s);
        out[(size_t)BB * 27 + b] = s;
    }
}

// ================= host launcher =================
extern "C" void kernel_launch(void* const* d_in, const int* in_sizes, int n_in,
                              void* d_out, int out_size, void* d_ws, size_t ws_size,
                              hipStream_t stream) {
    const float* x     = (const float*)d_in[0];
    const float* w_ih0 = (const float*)d_in[1];
    const float* w_hh0 = (const float*)d_in[2];
    const float* b_ih0 = (const float*)d_in[3];
    const float* b_hh0 = (const float*)d_in[4];
    const float* w_ih1 = (const float*)d_in[5];
    const float* w_hh1 = (const float*)d_in[6];
    const float* b_ih1 = (const float*)d_in[7];
    const float* b_hh1 = (const float*)d_in[8];
    const float* fc1w  = (const float*)d_in[9];
    const float* fc1b  = (const float*)d_in[10];
    const float* fc2w  = (const float*)d_in[11];
    const float* fc2b  = (const float*)d_in[12];

    char* ws = (char*)d_ws;
    unsigned short* pw_ih0 = (unsigned short*)(ws);             //  32768 B
    unsigned short* pw_hh0 = (unsigned short*)(ws + 32768);     // 131072 B
    unsigned short* pw_ih1 = (unsigned short*)(ws + 163840);    // 131072 B
    unsigned short* pw_hh1 = (unsigned short*)(ws + 294912);    // 131072 B
    float*          bias0  = (float*)(ws + 425984);             //   2048 B
    float*          bias1  = (float*)(ws + 428032);             //   2048 B
    float*          h2last = (float*)(ws + 430080);             // 131072 B

    // weight packing into WMMA B-fragment order (bf16)
    pack_b_kernel<<<(NTILES * 1 * 512 + 255) / 256, 256, 0, stream>>>(w_ih0, pw_ih0, II, 1);
    pack_b_kernel<<<(NTILES * 4 * 512 + 255) / 256, 256, 0, stream>>>(w_hh0, pw_hh0, HH, 4);
    pack_b_kernel<<<(NTILES * 4 * 512 + 255) / 256, 256, 0, stream>>>(w_ih1, pw_ih1, HH, 4);
    pack_b_kernel<<<(NTILES * 4 * 512 + 255) / 256, 256, 0, stream>>>(w_hh1, pw_hh1, HH, 4);
    bias_sum_kernel<<<2, 256, 0, stream>>>(b_ih0, b_hh0, bias0);
    bias_sum_kernel<<<2, 256, 0, stream>>>(b_ih1, b_hh1, bias1);

    // ~300KB dynamic LDS per workgroup (CDNA5 WGP has 320KB)
    (void)hipFuncSetAttribute((const void*)lstm_fused_kernel,
                              hipFuncAttributeMaxDynamicSharedMemorySize, 307456);
    lstm_fused_kernel<<<NWG, 256, 307456, stream>>>(x, pw_ih0, pw_hh0, pw_ih1, pw_hh1,
                                                    bias0, bias1, h2last);

    fc_kernel<<<BB, HH, 0, stream>>>(h2last, fc1w, fc1b, fc2w, fc2b, (float*)d_out);
}
